// HELMoEncoder_12292196401472
// MI455X (gfx1250) — matrix operand order repaired and checked by reading, not compiled, not run on hardware
//
#include <hip/hip_runtime.h>

// ---------------------------------------------------------------------------
// Problem constants (from reference)
// ---------------------------------------------------------------------------
#define BSZ  8192
#define IN   1024
#define HSZ  1024
#define DSZ  2048          // 2*H
#define NHEAD 16
#define HDIM 128           // D / NH
#define NCLS 7
#define SEQ  3

// GEMM tiling: 256x128 block tile, 8 waves, each computes a 64x64 sub-tile
#define TILE_M 256
#define TILE_N 128
#define TILE_K 32
#define KPAIRS (TILE_K / 2)   // 16 dwords per row (bf16 pairs)

#if __has_builtin(__builtin_amdgcn_global_load_async_to_lds_b128) && \
    __has_builtin(__builtin_amdgcn_s_wait_asynccnt)
#define HAVE_ASYNC_LDS 1
#else
#define HAVE_ASYNC_LDS 0
#warning "gfx1250 async-to-LDS builtins not available; using register-staging fallback"
#endif

typedef __bf16        v16bf __attribute__((ext_vector_type(16)));
typedef float         v8f   __attribute__((ext_vector_type(8)));
typedef unsigned int  v8u   __attribute__((ext_vector_type(8)));
typedef int           v4i   __attribute__((__vector_size__(16)));
typedef __attribute__((address_space(1))) v4i* gas_v4i;
typedef __attribute__((address_space(3))) v4i* las_v4i;

// RNE f32 -> bf16 pair packed into one dword (v_perm_b32 for the merge)
__device__ inline unsigned int pack_bf16(float a, float b) {
    unsigned int ua = __float_as_uint(a);
    unsigned int ub = __float_as_uint(b);
    ua += 0x7FFFu + ((ua >> 16) & 1u);
    ub += 0x7FFFu + ((ub >> 16) & 1u);
    return __builtin_amdgcn_perm(ub, ua, 0x07060302u);  // {ub.hi16, ua.hi16}
}

__device__ inline unsigned short bf16_of(float x) {
    unsigned int u = __float_as_uint(x);
    u += 0x7FFFu + ((u >> 16) & 1u);
    return (unsigned short)(u >> 16);
}

__device__ inline float bf16_to_f32(unsigned short h) {
    return __uint_as_float(((unsigned int)h) << 16);
}

// Load one 8-dword (16 bf16) fragment slice from LDS: two ds_load_b128.
__device__ inline v16bf load_frag8(const unsigned int* p) {
    const uint4 lo = *(const uint4*)(p);
    const uint4 hi = *(const uint4*)(p + 4);
    v8u u;
    u[0] = lo.x; u[1] = lo.y; u[2] = lo.z; u[3] = lo.w;
    u[4] = hi.x; u[5] = hi.y; u[6] = hi.z; u[7] = hi.w;
    return __builtin_bit_cast(v16bf, u);
}

#if HAVE_ASYNC_LDS
// One lane-wide async 16-byte copy: global -> LDS, tracked by ASYNCcnt.
__device__ inline void async_copy16(const unsigned short* g, unsigned int* l) {
    v4i* gg = (v4i*)g;   // const-cast + reinterpret in generic space
    v4i* ll = (v4i*)l;
    __builtin_amdgcn_global_load_async_to_lds_b128(
        (gas_v4i)gg, (las_v4i)ll, 0, 0);
}
#endif

// ---------------------------------------------------------------------------
// Bulk f32 -> bf16 conversion (RNE), 4 packed pairs (8 floats) per thread.
// ---------------------------------------------------------------------------
__global__ __launch_bounds__(256)
void f32_to_bf16(const float* __restrict__ src, unsigned int* __restrict__ dst,
                 size_t n_quads)
{
    const size_t i = (size_t)blockIdx.x * 256 + threadIdx.x;
    if (i >= n_quads) return;
    const float4 f0 = ((const float4*)src)[2 * i];
    const float4 f1 = ((const float4*)src)[2 * i + 1];
    uint4 o;
    o.x = pack_bf16(f0.x, f0.y);
    o.y = pack_bf16(f0.z, f0.w);
    o.z = pack_bf16(f1.x, f1.y);
    o.w = pack_bf16(f1.z, f1.w);
    ((uint4*)dst)[i] = o;
}

// ---------------------------------------------------------------------------
// C[M,N] = scale * (A[M,K] @ W[N,K]^T), A/W bf16 in global, f32 accumulate
// via v_wmma_f32_16x16x32_bf16. Double-buffered LDS fed by the gfx1250
// async-to-LDS DMA path (ASYNCcnt overlaps the copies with the WMMAs).
//
// LDS layouts (per 32-K tile, one dword = one bf16 pair):
//   sB: plain row-major [n][kpair]; lane (n = lane&15) reads contiguous
//       8 dwords at (lane>=16)*8  (contiguous K: 0..15 / 16..31, same order as sA).
//   sA: plain row-major [m][kpair] too: dword positions hold pairs
//       [0..15] in order; lane group 0 reads pos 0..7 =
//       K 0..15, group 1 reads pos 8..15 = K 16..31 —
//       the same contiguous k order as sB (not the ISA table order). Chunks are
//       copied to per-chunk LDS destinations +0,+4,+8,+12 (no permutation).
//
// Block tile 256x128, 8 waves in 4x2, each wave a 64x64 sub-tile:
// 4 A-frags + 4 B-frags -> 16 WMMAs per 32-K step.
// ---------------------------------------------------------------------------
__global__ __launch_bounds__(256, 1)
void gemm_bf16_wmma(const unsigned short* __restrict__ A, int lda,
                    const unsigned short* __restrict__ W, int ldw,
                    float* __restrict__ C, int ldc,
                    int M, int N, int K, float scale)
{
    __shared__ alignas(16) unsigned int sA[2][TILE_M * KPAIRS];  // 2 x 16 KB
    __shared__ alignas(16) unsigned int sB[2][TILE_N * KPAIRS];  // 2 x  8 KB

    const int tid  = threadIdx.x;
    const int lane = tid & 31;
    const int wave = tid >> 5;
    const int wm   = wave & 3;   // 4 row-slabs of 64
    const int wn   = wave >> 2;  // 2 col-slabs of 64
    const int bm   = blockIdx.y * TILE_M;
    const int bn   = blockIdx.x * TILE_N;

    v8f acc[4][4];
#pragma unroll
    for (int mi = 0; mi < 4; ++mi)
#pragma unroll
        for (int ni = 0; ni < 4; ++ni)
            acc[mi][ni] = v8f{0.f, 0.f, 0.f, 0.f, 0.f, 0.f, 0.f, 0.f};

    // A loader: 1 thread per row, 4x16B chunks (in-order positions)
    // B loader: 2 threads per row, 2x16B chunks each
    const int ld_rowB = tid >> 1;
    const int ld_half = tid & 1;
    const unsigned short* gA = A + (size_t)(bm + tid) * lda;
    const unsigned short* gW = W + (size_t)(bn + ld_rowB) * ldw + (ld_half << 4);
    const int aBase = tid * KPAIRS;
    const int bOff  = ld_rowB * KPAIRS + (ld_half << 3);
    const int aPos0 = 0, aPos1 = 4, aPos2 = 8, aPos3 = 12;  // chunk LDS offsets, in k order

    const int nk = K / TILE_K;

    // ---- stage tile 0 into LDS buffer 0 ----
#if HAVE_ASYNC_LDS
    async_copy16(gA,      &sA[0][aBase + aPos0]);
    async_copy16(gA + 8,  &sA[0][aBase + aPos1]);
    async_copy16(gA + 16, &sA[0][aBase + aPos2]);
    async_copy16(gA + 24, &sA[0][aBase + aPos3]);
    async_copy16(gW,      &sB[0][bOff]);
    async_copy16(gW + 8,  &sB[0][bOff + 4]);
    __builtin_amdgcn_s_wait_asynccnt(0);
#else
    {
        const uint4 a0 = ((const uint4*)gA)[0];
        const uint4 a1 = ((const uint4*)gA)[1];
        const uint4 a2 = ((const uint4*)gA)[2];
        const uint4 a3 = ((const uint4*)gA)[3];
        const uint4 b0 = ((const uint4*)gW)[0];
        const uint4 b1 = ((const uint4*)gW)[1];
        *(uint4*)&sA[0][aBase + aPos0] = a0;
        *(uint4*)&sA[0][aBase + aPos1] = a1;
        *(uint4*)&sA[0][aBase + aPos2] = a2;
        *(uint4*)&sA[0][aBase + aPos3] = a3;
        *(uint4*)&sB[0][bOff]     = b0;
        *(uint4*)&sB[0][bOff + 4] = b1;
    }
#endif

    int cur = 0;
    for (int ki = 0; ki < nk; ++ki) {
        __syncthreads();
        const bool has_next = (ki + 1) < nk;
        const int  nxt      = cur ^ 1;

#if HAVE_ASYNC_LDS
        if (has_next) {   // async copies run during the WMMAs below
            const unsigned short* pa = gA + (size_t)(ki + 1) * TILE_K;
            const unsigned short* pw = gW + (size_t)(ki + 1) * TILE_K;
            async_copy16(pa,      &sA[nxt][aBase + aPos0]);
            async_copy16(pa + 8,  &sA[nxt][aBase + aPos1]);
            async_copy16(pa + 16, &sA[nxt][aBase + aPos2]);
            async_copy16(pa + 24, &sA[nxt][aBase + aPos3]);
            async_copy16(pw,      &sB[nxt][bOff]);
            async_copy16(pw + 8,  &sB[nxt][bOff + 4]);
        }
#endif
        if (ki + 2 < nk) {   // global_prefetch_b8 two tiles ahead
            __builtin_prefetch(gA + (size_t)(ki + 2) * TILE_K, 0, 0);
            __builtin_prefetch(gW + (size_t)(ki + 2) * TILE_K, 0, 0);
        }

        // ---- compute on current buffer: 8 frag loads, 16 WMMAs ----
        const int fsel = (lane >> 4) << 3;   // 0 or 8 dwords into the row
        v16bf bf[4];
#pragma unroll
        for (int ni = 0; ni < 4; ++ni)
            bf[ni] = load_frag8(&sB[cur][(wn * 64 + ni * 16 + (lane & 15)) * KPAIRS + fsel]);
#pragma unroll
        for (int mi = 0; mi < 4; ++mi) {
            v16bf af = load_frag8(&sA[cur][(wm * 64 + mi * 16 + (lane & 15)) * KPAIRS + fsel]);
#pragma unroll
            for (int ni = 0; ni < 4; ++ni)
                acc[mi][ni] = __builtin_amdgcn_wmma_f32_16x16x32_bf16(
                    false, af, false, bf[ni], (short)0, acc[mi][ni], false, false);
        }

#if HAVE_ASYNC_LDS
        if (has_next) __builtin_amdgcn_s_wait_asynccnt(0);  // copies done before next barrier
#else
        if (has_next) {   // short-live-range copy after the compute
            const unsigned short* pa = gA + (size_t)(ki + 1) * TILE_K;
            const unsigned short* pw = gW + (size_t)(ki + 1) * TILE_K;
            const uint4 a0 = ((const uint4*)pa)[0];
            const uint4 a1 = ((const uint4*)pa)[1];
            const uint4 a2 = ((const uint4*)pa)[2];
            const uint4 a3 = ((const uint4*)pa)[3];
            const uint4 b0 = ((const uint4*)pw)[0];
            const uint4 b1 = ((const uint4*)pw)[1];
            *(uint4*)&sA[nxt][aBase + aPos0] = a0;
            *(uint4*)&sA[nxt][aBase + aPos1] = a1;
            *(uint4*)&sA[nxt][aBase + aPos2] = a2;
            *(uint4*)&sA[nxt][aBase + aPos3] = a3;
            *(uint4*)&sB[nxt][bOff]     = b0;
            *(uint4*)&sB[nxt][bOff + 4] = b1;
        }
#endif
        cur ^= 1;
    }

    // store: C/D layout — VGPR r, lane l: row = r + (l>=16)*8, col = l&15
#pragma unroll
    for (int mi = 0; mi < 4; ++mi) {
#pragma unroll
        for (int ni = 0; ni < 4; ++ni) {
            const int col  = bn + wn * 64 + ni * 16 + (lane & 15);
            const int row0 = bm + wm * 64 + mi * 16 + ((lane >> 4) << 3);
#pragma unroll
            for (int r = 0; r < 8; ++r)
                C[(size_t)(row0 + r) * ldc + col] = acc[mi][ni][r] * scale;
        }
    }
}

// ---------------------------------------------------------------------------
// Fused PyTorch GRU cell (gate order r,z,n). gh == nullptr => h_prev = 0.
// ---------------------------------------------------------------------------
__global__ __launch_bounds__(256)
void gru_gate(const float* __restrict__ gi, const float* __restrict__ gh,
              const float* __restrict__ b_ih, const float* __restrict__ b_hh,
              const unsigned short* __restrict__ hprev, int hprev_stride,
              unsigned short* __restrict__ hout, int hout_stride)
{
    const size_t idx = (size_t)blockIdx.x * 256 + threadIdx.x;
    if (idx >= (size_t)BSZ * HSZ) return;
    const int b = (int)(idx / HSZ);
    const int j = (int)(idx % HSZ);
    const size_t g = (size_t)b * 3 * HSZ + j;

    float ir = gi[g]            + b_ih[j];
    float iz = gi[g + HSZ]      + b_ih[HSZ + j];
    float in = gi[g + 2 * HSZ]  + b_ih[2 * HSZ + j];
    float hr = b_hh[j];
    float hz = b_hh[HSZ + j];
    float hn = b_hh[2 * HSZ + j];
    if (gh) {
        hr += gh[g];
        hz += gh[g + HSZ];
        hn += gh[g + 2 * HSZ];
    }
    const float r = 1.f / (1.f + __expf(-(ir + hr)));
    const float z = 1.f / (1.f + __expf(-(iz + hz)));
    const float n = tanhf(in + r * hn);
    const float hp = hprev ? bf16_to_f32(hprev[(size_t)b * hprev_stride + j]) : 0.f;
    hout[(size_t)b * hout_stride + j] = bf16_of((1.f - z) * n + z * hp);
}

// ---------------------------------------------------------------------------
// Tiny attention core: one wave per (batch, head). S=3, HD=128 (4 floats/lane).
// Context written as packed bf16 pairs (feeds the O-projection GEMM).
// ---------------------------------------------------------------------------
__global__ __launch_bounds__(256)
void attn_small(const float* __restrict__ q, const float* __restrict__ k,
                const float* __restrict__ v, unsigned int* __restrict__ ctx)
{
    const int unit = blockIdx.x * 8 + (threadIdx.x >> 5);   // b*NH + nh
    if (unit >= BSZ * NHEAD) return;
    const int lane = threadIdx.x & 31;
    const int b  = unit >> 4;
    const int nh = unit & 15;
    const size_t base = (size_t)b * SEQ * DSZ + nh * HDIM + lane * 4;

    float4 qv[SEQ], kv[SEQ], vv[SEQ];
#pragma unroll
    for (int s = 0; s < SEQ; ++s) {
        qv[s] = *(const float4*)(q + base + (size_t)s * DSZ);
        kv[s] = *(const float4*)(k + base + (size_t)s * DSZ);
        vv[s] = *(const float4*)(v + base + (size_t)s * DSZ);
    }

    float lg[SEQ][SEQ];
#pragma unroll
    for (int qi = 0; qi < SEQ; ++qi)
#pragma unroll
        for (int ki = 0; ki < SEQ; ++ki) {
            float p = qv[qi].x * kv[ki].x + qv[qi].y * kv[ki].y +
                      qv[qi].z * kv[ki].z + qv[qi].w * kv[ki].w;
#pragma unroll
            for (int off = 16; off > 0; off >>= 1)
                p += __shfl_xor(p, off, 32);
            lg[qi][ki] = p;
        }

#pragma unroll
    for (int qi = 0; qi < SEQ; ++qi) {
        const float m  = fmaxf(fmaxf(lg[qi][0], lg[qi][1]), lg[qi][2]);
        const float e0 = __expf(lg[qi][0] - m);
        const float e1 = __expf(lg[qi][1] - m);
        const float e2 = __expf(lg[qi][2] - m);
        const float inv = 1.f / (e0 + e1 + e2);
        float4 c;
        c.x = (e0 * vv[0].x + e1 * vv[1].x + e2 * vv[2].x) * inv;
        c.y = (e0 * vv[0].y + e1 * vv[1].y + e2 * vv[2].y) * inv;
        c.z = (e0 * vv[0].z + e1 * vv[1].z + e2 * vv[2].z) * inv;
        c.w = (e0 * vv[0].w + e1 * vv[1].w + e2 * vv[2].w) * inv;
        uint2 o;
        o.x = pack_bf16(c.x, c.y);
        o.y = pack_bf16(c.z, c.w);
        *(uint2*)(ctx + (base >> 1) + (size_t)qi * (DSZ / 2)) = o;
    }
}

// hsum[b][d] = sum_s att[b][s][d]
__global__ __launch_bounds__(256)
void seq_sum(const float* __restrict__ att, float* __restrict__ hsum)
{
    const size_t idx = (size_t)blockIdx.x * 256 + threadIdx.x;
    if (idx >= (size_t)BSZ * DSZ) return;
    const size_t b = idx / DSZ;
    const size_t d = idx % DSZ;
    const size_t base = b * SEQ * DSZ + d;
    hsum[idx] = att[base] + att[base + DSZ] + att[base + 2 * DSZ];
}

// One block per batch row: 7 dots of length D, then softmax; writes (o, p).
__global__ __launch_bounds__(256)
void classifier_softmax(const float* __restrict__ h, const float* __restrict__ Wout,
                        const float* __restrict__ bout, float* __restrict__ out)
{
    __shared__ float red[256 * 8];
    const int b = blockIdx.x;
    const int t = threadIdx.x;
    float acc[NCLS];
#pragma unroll
    for (int c = 0; c < NCLS; ++c) acc[c] = 0.f;

    for (int d = t; d < DSZ; d += 256) {
        const float hv = h[(size_t)b * DSZ + d];
#pragma unroll
        for (int c = 0; c < NCLS; ++c)
            acc[c] += hv * Wout[(size_t)c * DSZ + d];
    }
#pragma unroll
    for (int c = 0; c < NCLS; ++c) red[t * 8 + c] = acc[c];
    __syncthreads();
    for (int s = 128; s > 0; s >>= 1) {
        if (t < s)
#pragma unroll
            for (int c = 0; c < NCLS; ++c) red[t * 8 + c] += red[(t + s) * 8 + c];
        __syncthreads();
    }
    if (t == 0) {
        float o[NCLS], mx = -3.4e38f;
#pragma unroll
        for (int c = 0; c < NCLS; ++c) { o[c] = red[c] + bout[c]; mx = fmaxf(mx, o[c]); }
        float e[NCLS], sum = 0.f;
#pragma unroll
        for (int c = 0; c < NCLS; ++c) { e[c] = __expf(o[c] - mx); sum += e[c]; }
        const float inv = 1.f / sum;
#pragma unroll
        for (int c = 0; c < NCLS; ++c) {
            out[(size_t)b * NCLS + c] = o[c];
            out[(size_t)BSZ * NCLS + (size_t)b * NCLS + c] = e[c] * inv;
        }
    }
}

// ---------------------------------------------------------------------------
// Host-side orchestration
// ---------------------------------------------------------------------------
static inline void launch_gemm(const unsigned short* A, int lda,
                               const unsigned short* W, int ldw,
                               float* C, int ldc, int M, int N, int K,
                               float scale, hipStream_t stream)
{
    dim3 grid(N / TILE_N, M / TILE_M);
    gemm_bf16_wmma<<<grid, 256, 0, stream>>>(A, lda, W, ldw, C, ldc, M, N, K, scale);
}

static inline void launch_cvt(const float* src, unsigned short* dst, size_t elems,
                              hipStream_t stream)
{
    const size_t quads = elems / 8;   // 8 floats per thread
    f32_to_bf16<<<(unsigned)((quads + 255) / 256), 256, 0, stream>>>(
        src, (unsigned int*)dst, quads);
}

extern "C" void kernel_launch(void* const* d_in, const int* in_sizes, int n_in,
                              void* d_out, int out_size, void* d_ws, size_t ws_size,
                              hipStream_t stream)
{
    const float* x[3]    = {(const float*)d_in[0], (const float*)d_in[1], (const float*)d_in[2]};
    const float* W_ih_f  = (const float*)d_in[3];
    const float* W_hh_f  = (const float*)d_in[4];
    const float* b_ih_f  = (const float*)d_in[5];
    const float* b_hh_f  = (const float*)d_in[6];
    const float* W_ih_b  = (const float*)d_in[7];
    const float* W_hh_b  = (const float*)d_in[8];
    const float* b_ih_b  = (const float*)d_in[9];
    const float* b_hh_b  = (const float*)d_in[10];
    const float* Wq      = (const float*)d_in[11];
    const float* Wk      = (const float*)d_in[12];
    const float* Wv      = (const float*)d_in[13];
    const float* Wo      = (const float*)d_in[14];
    const float* W_out   = (const float*)d_in[15];
    const float* b_out   = (const float*)d_in[16];
    float* out = (float*)d_out;

    // ---- workspace layout ----
    const size_t HS_N = (size_t)BSZ * SEQ * DSZ;   // 50331648
    const size_t GI_N = (size_t)BSZ * 3 * HSZ;     // 25165824
    const size_t XI_N = (size_t)BSZ * IN;          // 8388608
    const size_t WI_N = (size_t)3 * HSZ * IN;      // 3145728  (W_ih / W_hh)
    const size_t WD_N = (size_t)DSZ * DSZ;         // 4194304  (Wq/Wk/Wv/Wo)

    float* ws = (float*)d_ws;
    float* qb = ws;                 // f32 Q
    float* kb = qb + HS_N;          // f32 K, reused for attention output
    float* vb = kb + HS_N;          // f32 V
    float* gi = vb + HS_N;          // f32 GRU input-gates; reused for hsum
    float* gh = gi + GI_N;          // f32 GRU hidden-gates

    unsigned short* bfb   = (unsigned short*)(gh + GI_N);
    unsigned short* hs_bf = bfb;                 // bf16 hs (B,3,D)
    unsigned short* ctx_bf= hs_bf + HS_N;        // bf16 attention context
    unsigned short* xbf0  = ctx_bf + HS_N;
    unsigned short* xbf1  = xbf0 + XI_N;
    unsigned short* xbf2  = xbf1 + XI_N;
    unsigned short* wihf  = xbf2 + XI_N;
    unsigned short* whhf  = wihf + WI_N;
    unsigned short* wihb  = whhf + WI_N;
    unsigned short* whhb  = wihb + WI_N;
    unsigned short* wqb   = whhb + WI_N;
    unsigned short* wkb   = wqb + WD_N;
    unsigned short* wvb   = wkb + WD_N;
    unsigned short* wob   = wvb + WD_N;
    unsigned short* xbf[3] = {xbf0, xbf1, xbf2};

    float* att  = kb;   // attention output reuses K buffer
    float* hsum = gi;   // head sum reuses gi buffer

    const int gate_blocks = (int)(((size_t)BSZ * HSZ + 255) / 256);
    const float qscale = 0.08838834764831845f;   // HD^-0.5

    // ---- one-time bf16 conversions (weights + inputs) ----
    launch_cvt(x[0], xbf0, XI_N, stream);
    launch_cvt(x[1], xbf1, XI_N, stream);
    launch_cvt(x[2], xbf2, XI_N, stream);
    launch_cvt(W_ih_f, wihf, WI_N, stream);
    launch_cvt(W_hh_f, whhf, WI_N, stream);
    launch_cvt(W_ih_b, wihb, WI_N, stream);
    launch_cvt(W_hh_b, whhb, WI_N, stream);
    launch_cvt(Wq, wqb, WD_N, stream);
    launch_cvt(Wk, wkb, WD_N, stream);
    launch_cvt(Wv, wvb, WD_N, stream);
    launch_cvt(Wo, wob, WD_N, stream);

    // ---------------- forward GRU direction (cols 0..H-1 of hs) -------------
    for (int t = 0; t < SEQ; ++t) {
        launch_gemm(xbf[t], IN, wihf, IN, gi, 3 * HSZ, BSZ, 3 * HSZ, IN, 1.f, stream);
        if (t == 0) {
            gru_gate<<<gate_blocks, 256, 0, stream>>>(
                gi, nullptr, b_ih_f, b_hh_f, nullptr, 0,
                hs_bf + (size_t)t * DSZ, SEQ * DSZ);
        } else {
            const unsigned short* hprev = hs_bf + (size_t)(t - 1) * DSZ;
            launch_gemm(hprev, SEQ * DSZ, whhf, HSZ, gh, 3 * HSZ,
                        BSZ, 3 * HSZ, HSZ, 1.f, stream);
            gru_gate<<<gate_blocks, 256, 0, stream>>>(
                gi, gh, b_ih_f, b_hh_f, hprev, SEQ * DSZ,
                hs_bf + (size_t)t * DSZ, SEQ * DSZ);
        }
    }
    // ---------------- backward GRU direction (cols H..2H-1) -----------------
    for (int p = 0; p < SEQ; ++p) {
        const int s = SEQ - 1 - p;   // sequence position processed at step p
        launch_gemm(xbf[s], IN, wihb, IN, gi, 3 * HSZ, BSZ, 3 * HSZ, IN, 1.f, stream);
        if (p == 0) {
            gru_gate<<<gate_blocks, 256, 0, stream>>>(
                gi, nullptr, b_ih_b, b_hh_b, nullptr, 0,
                hs_bf + (size_t)s * DSZ + HSZ, SEQ * DSZ);
        } else {
            const unsigned short* hprev = hs_bf + (size_t)(s + 1) * DSZ + HSZ;
            launch_gemm(hprev, SEQ * DSZ, whhb, HSZ, gh, 3 * HSZ,
                        BSZ, 3 * HSZ, HSZ, 1.f, stream);
            gru_gate<<<gate_blocks, 256, 0, stream>>>(
                gi, gh, b_ih_b, b_hh_b, hprev, SEQ * DSZ,
                hs_bf + (size_t)s * DSZ + HSZ, SEQ * DSZ);
        }
    }

    // ---------------- multi-head attention ----------------------------------
    const int Mqkv = BSZ * SEQ;   // 24576
    launch_gemm(hs_bf, DSZ, wqb, DSZ, qb, DSZ, Mqkv, DSZ, DSZ, qscale, stream);
    launch_gemm(hs_bf, DSZ, wkb, DSZ, kb, DSZ, Mqkv, DSZ, DSZ, 1.f, stream);
    launch_gemm(hs_bf, DSZ, wvb, DSZ, vb, DSZ, Mqkv, DSZ, DSZ, 1.f, stream);

    attn_small<<<(BSZ * NHEAD) / 8, 256, 0, stream>>>(
        qb, kb, vb, (unsigned int*)ctx_bf);

    launch_gemm(ctx_bf, DSZ, wob, DSZ, att, DSZ, Mqkv, DSZ, DSZ, 1.f, stream);

    // ---------------- head: sum over sequence + classifier + softmax --------
    seq_sum<<<(int)(((size_t)BSZ * DSZ + 255) / 256), 256, 0, stream>>>(att, hsum);
    classifier_softmax<<<BSZ, 256, 0, stream>>>(hsum, W_out, b_out, out);
}
